// ChannelAttention_25323127177327
// MI455X (gfx1250) — matrix operand order repaired and checked by reading, not compiled, not run on hardware
//
#include <hip/hip_runtime.h>
#include <stdint.h>

// CDNA5 / gfx1250, wave32. bf16 WMMA 16x16x32 f32-acc + async global->LDS staging.
typedef __attribute__((ext_vector_type(16))) __bf16 v16bf;
typedef __attribute__((ext_vector_type(8)))  float  v8f;

union FragBF { uint32_t u[8]; v16bf v; };
union Pack4  { uint16_t h[4]; uint64_t u; };

#define B_DIM 32
#define C_DIM 1024
#define D_DIM 784
#define D_PAD 800         // 25 * 32
#define BT_STRIDE 40      // LDS row stride (elems): 80B rows, 16B aligned, conflict-free
#define MT_STRIDE 40

static __device__ __forceinline__ uint16_t f2bf(float f) {
  uint32_t u = __builtin_bit_cast(uint32_t, f);
  u += 0x7FFFu + ((u >> 16) & 1u);     // round-to-nearest-even
  return (uint16_t)(u >> 16);
}

// CDNA5 async copy: 16B global -> LDS per lane, tracked by ASYNCcnt.
static __device__ __forceinline__ void async_copy16(void* lds_dst, const uint16_t* g) {
  uint32_t lds_off = (uint32_t)(uintptr_t)lds_dst;   // generic ptr low 32 = LDS offset
  asm volatile("global_load_async_to_lds_b128 %0, %1, off"
               :: "v"(lds_off), "v"(g) : "memory");
}
static __device__ __forceinline__ void wait_async0() {
  asm volatile("s_wait_asynccnt 0x0" ::: "memory");
}

// ---- Pass 0: fp32 -> bf16; write Mbf[b][c][800] and transposed Mt[b][800][1024]
__global__ void __launch_bounds__(256)
cvt_kernel(const float* __restrict__ x,
           uint16_t* __restrict__ mbf, uint16_t* __restrict__ mt) {
  __shared__ uint16_t tl[32][33];
  const int b   = blockIdx.z;
  const int j0  = blockIdx.y * 32;
  const int d0  = blockIdx.x * 32;
  const int tid = threadIdx.x;
  {
    const int jj = tid >> 3;
    const int dd = (tid & 7) * 4;
    Pack4 p;
#pragma unroll
    for (int i = 0; i < 4; ++i) {
      const int d = d0 + dd + i;
      float v = (d < D_DIM) ? x[((size_t)b * C_DIM + j0 + jj) * D_DIM + d] : 0.0f;
      p.h[i] = f2bf(v);
      tl[jj][dd + i] = p.h[i];
    }
    *(uint64_t*)(mbf + ((size_t)b * C_DIM + j0 + jj) * D_PAD + d0 + dd) = p.u;
  }
  __syncthreads();
  {
    const int dd = tid >> 3;
    const int jj = (tid & 7) * 4;
    Pack4 p;
#pragma unroll
    for (int i = 0; i < 4; ++i) p.h[i] = tl[jj + i][dd];
    *(uint64_t*)(mt + ((size_t)b * D_PAD + d0 + dd) * C_DIM + j0 + jj) = p.u;
  }
}

// ---- Pass 1: E = exp(scale * M M^T) (bf16) + per-row sums (f32)
// grid = (8 row-tiles, 32 batches); block 256 (8 waves x 16 rows)
// Nested loops (clean acc codegen) + double-buffered async staging, 1 barrier/K-step.
__global__ void __launch_bounds__(256)
scores_kernel(const uint16_t* __restrict__ mbf,
              uint16_t* __restrict__ e, float* __restrict__ rowsum) {
  __shared__ uint16_t bt[2][64 * BT_STRIDE];        // j-tile 64 x k 32 (x2 bufs)
  const int b    = blockIdx.y;
  const int tid  = threadIdx.x;
  const int wave = tid >> 5;
  const int lane = tid & 31;
  const int hi   = (lane >> 4) & 1;
  const int ln   = lane & 15;
  const int rowbase = blockIdx.x * 128 + wave * 16;

  const uint16_t* mb   = mbf + (size_t)b * C_DIM * D_PAD;
  const uint16_t* arow = mb + (size_t)(rowbase + ln) * D_PAD;

  const int sjj = tid >> 2;                         // staging: 64 rows x 4 chunks
  const int skk = (tid & 3) * 8;

  float sums[8];
#pragma unroll
  for (int v = 0; v < 8; ++v) sums[v] = 0.0f;

  int it = 0;
  // prologue: stage tile (jb=0, ks=0) into buf 0
  async_copy16(&bt[0][sjj * BT_STRIDE + skk], mb + (size_t)sjj * D_PAD + skk);

  for (int jb = 0; jb < 16; ++jb) {                 // 16 j-blocks of 64 cols
    const int j0 = jb * 64;
    v8f acc[4];
#pragma unroll
    for (int t = 0; t < 4; ++t)
#pragma unroll
      for (int v = 0; v < 8; ++v) acc[t][v] = 0.0f;

    for (int ks = 0; ks < 25; ++ks, ++it) {         // K = 800 / 32
      const int k0 = ks * 32;
      const int cur = it & 1;

      wait_async0();                                // own staged writes done
      __syncthreads();                              // buf[cur] published

      int jbn = jb, ksn = ks + 1;                   // prefetch next tile
      if (ksn == 25) { ksn = 0; jbn = jb + 1; }
      if (jbn < 16)
        async_copy16(&bt[cur ^ 1][sjj * BT_STRIDE + skk],
                     mb + (size_t)(jbn * 64 + sjj) * D_PAD + ksn * 32 + skk);

      FragBF a;                                     // A: this wave's 16 rows
#pragma unroll
      for (int r = 0; r < 4; ++r)
        a.u[r]     = *(const uint32_t*)(arow + k0 + hi * 16 + 2 * r);
#pragma unroll
      for (int r = 0; r < 4; ++r)
        a.u[4 + r] = *(const uint32_t*)(arow + k0 + hi * 16 + 8 + 2 * r);

#pragma unroll
      for (int t = 0; t < 4; ++t) {
        FragBF bf;
        const int jcol = t * 16 + ln;
#pragma unroll
        for (int r = 0; r < 8; ++r)
          bf.u[r] = *(const uint32_t*)&bt[cur][jcol * BT_STRIDE + hi * 16 + 2 * r];
        acc[t] = __builtin_amdgcn_wmma_f32_16x16x32_bf16(
            false, a.v, false, bf.v, (short)0, acc[t], false, false);
      }
    }

    // epilogue: E = exp(scale*S); overlaps with already-issued next-jb staging
    const float scale = 0.03125f;                   // 1024^-0.5
#pragma unroll
    for (int t = 0; t < 4; ++t) {
      const int jcol = j0 + t * 16 + ln;
#pragma unroll
      for (int v = 0; v < 8; ++v) {
        const int row = rowbase + hi * 8 + v;
        float ev = __expf(acc[t][v] * scale);       // max score ~24.5: safe
        sums[v] += ev;
        e[((size_t)b * C_DIM + row) * C_DIM + jcol] = f2bf(ev);
      }
    }
  }

#pragma unroll
  for (int v = 0; v < 8; ++v) {                     // reduce within 16-lane halves
    float s = sums[v];
    s += __shfl_xor(s, 1);
    s += __shfl_xor(s, 2);
    s += __shfl_xor(s, 4);
    s += __shfl_xor(s, 8);
    sums[v] = s;
  }
  if (ln == 0) {
#pragma unroll
    for (int v = 0; v < 8; ++v)
      rowsum[(size_t)b * C_DIM + rowbase + hi * 8 + v] = sums[v];
  }
}

// ---- Pass 2: out = x + (E @ M) / rowsum
// B operand from transposed Mt: K-pairs contiguous -> async b128 staging.
__global__ void __launch_bounds__(256)
out_kernel(const uint16_t* __restrict__ e,
           const uint16_t* __restrict__ mt,
           const float* __restrict__ rowsum,
           const float* __restrict__ x, float* __restrict__ out) {
  __shared__ uint16_t mtile[2][80 * MT_STRIDE];     // d-tile 80 x j 32 (x2 bufs)
  const int b    = blockIdx.y;
  const int tid  = threadIdx.x;
  const int wave = tid >> 5;
  const int lane = tid & 31;
  const int hi   = (lane >> 4) & 1;
  const int ln   = lane & 15;
  const int rowbase = blockIdx.x * 128 + wave * 16;

  const uint16_t* mtg  = mt + (size_t)b * D_PAD * C_DIM;
  const uint16_t* erow = e + ((size_t)b * C_DIM + rowbase + ln) * C_DIM;

  float rinv[8];
#pragma unroll
  for (int v = 0; v < 8; ++v)
    rinv[v] = 1.0f / rowsum[(size_t)b * C_DIM + rowbase + hi * 8 + v];

  // staging tasks: 80 rows x 4 chunks of 16B = 320; thread does tid and tid+256
  const int s0dd = tid >> 2;          const int s0kk = (tid & 3) * 8;
  const int s1dd = (tid + 256) >> 2;  const int s1kk = ((tid + 256) & 3) * 8;
  const bool s1on = tid < 64;

  int it = 0;
  // prologue: stage tile (db=0, ks=0) into buf 0
  async_copy16(&mtile[0][s0dd * MT_STRIDE + s0kk], mtg + (size_t)s0dd * C_DIM + s0kk);
  if (s1on)
    async_copy16(&mtile[0][s1dd * MT_STRIDE + s1kk], mtg + (size_t)s1dd * C_DIM + s1kk);

  for (int db = 0; db < 10; ++db) {                 // 10 d-blocks of 80
    const int d0 = db * 80;
    v8f acc[5];
#pragma unroll
    for (int t = 0; t < 5; ++t)
#pragma unroll
      for (int v = 0; v < 8; ++v) acc[t][v] = 0.0f;

    for (int ks = 0; ks < 32; ++ks, ++it) {         // K = 1024 / 32 (over j)
      const int j0 = ks * 32;
      const int cur = it & 1;

      wait_async0();
      __syncthreads();

      int dbn = db, ksn = ks + 1;                   // prefetch next tile
      if (ksn == 32) { ksn = 0; dbn = db + 1; }
      if (dbn < 10) {
        const uint16_t* src = mtg + (size_t)(dbn * 80) * C_DIM + ksn * 32;
        async_copy16(&mtile[cur ^ 1][s0dd * MT_STRIDE + s0kk],
                     src + (size_t)s0dd * C_DIM + s0kk);
        if (s1on)
          async_copy16(&mtile[cur ^ 1][s1dd * MT_STRIDE + s1kk],
                       src + (size_t)s1dd * C_DIM + s1kk);
      }

      FragBF a;                                     // A: E rows, contiguous j
#pragma unroll
      for (int r = 0; r < 4; ++r)
        a.u[r]     = *(const uint32_t*)(erow + j0 + hi * 16 + 2 * r);
#pragma unroll
      for (int r = 0; r < 4; ++r)
        a.u[4 + r] = *(const uint32_t*)(erow + j0 + hi * 16 + 8 + 2 * r);

#pragma unroll
      for (int t = 0; t < 5; ++t) {
        FragBF bf;                                  // B[k][n] = Mt[d0+n][j0+k]
        const int drow = t * 16 + ln;
#pragma unroll
        for (int r = 0; r < 8; ++r)
          bf.u[r] = *(const uint32_t*)&mtile[cur][drow * MT_STRIDE + hi * 16 + 2 * r];
        acc[t] = __builtin_amdgcn_wmma_f32_16x16x32_bf16(
            false, a.v, false, bf.v, (short)0, acc[t], false, false);
      }
    }

    // epilogue: residual add + normalize, drop padded cols
#pragma unroll
    for (int t = 0; t < 5; ++t) {
      const int dcol = d0 + t * 16 + ln;
      if (dcol < D_DIM) {
#pragma unroll
        for (int v = 0; v < 8; ++v) {
          const int row = rowbase + hi * 8 + v;
          const size_t idx = ((size_t)b * C_DIM + row) * D_DIM + dcol;
          out[idx] = x[idx] + acc[t][v] * rinv[v];
        }
      }
    }
  }
}

extern "C" void kernel_launch(void* const* d_in, const int* in_sizes, int n_in,
                              void* d_out, int out_size, void* d_ws, size_t ws_size,
                              hipStream_t stream) {
  const float* x = (const float*)d_in[0];
  float* out = (float*)d_out;

  // ws: Mbf 52.4MB | Mt 52.4MB | E 67.1MB | rowsum 128KB  (~172MB, fits 192MB L2)
  uint16_t* mbf = (uint16_t*)d_ws;
  uint16_t* mt  = mbf + (size_t)B_DIM * C_DIM * D_PAD;
  uint16_t* e   = mt  + (size_t)B_DIM * D_PAD * C_DIM;
  float* rsum   = (float*)(e + (size_t)B_DIM * C_DIM * C_DIM);

  cvt_kernel   <<<dim3(25, 32, B_DIM), dim3(256), 0, stream>>>(x, mbf, mt);
  scores_kernel<<<dim3(8, B_DIM),      dim3(256), 0, stream>>>(mbf, e, rsum);
  out_kernel   <<<dim3(8, B_DIM),      dim3(256), 0, stream>>>(e, mt, rsum, x, out);
}
